// CNV_w1a1_DEEP_Resnet_V3_NOMAXPOOL_23021024707486
// MI455X (gfx1250) — 16-bit and 8-bit matrix operand order repaired and checked by reading, compiled, not run on hardware
//
#include <hip/hip_runtime.h>
#include <hip/hip_bf16.h>
#include <cstdint>

typedef __attribute__((ext_vector_type(8)))  int      v8i;
typedef __attribute__((ext_vector_type(8)))  float    v8f;
typedef __attribute__((ext_vector_type(16))) _Float16 v16h;
typedef __attribute__((ext_vector_type(8)))  _Float16 v8h;

#define BN_EPS 1e-4f

template <int CIN, int KK, int NT>
__global__ __launch_bounds__(128) void binconv_t(
    const int8_t* __restrict__ in, int8_t* __restrict__ out,
    const int* __restrict__ wp,
    const float* __restrict__ bg, const float* __restrict__ bb,
    const float* __restrict__ bm, const float* __restrict__ bv,
    int H, int W, int Cout, int stride, int pad, int Ho, int Wo)
{
  constexpr int K = CIN * KK * KK;
  constexpr int kCh = (K + 63) / 64;
  int lane = threadIdx.x & 31;
  int wave = threadIdx.x >> 5;
  long long M = 32ll * Ho * Wo;
  long long mTiles = M >> 4;
  int nGrp = (Cout >> 4) / NT;
  long long tile = (long long)blockIdx.x * 4 + wave;
  if (tile >= mTiles * nGrp) return;
  int tg = (int)(tile % nGrp);
  long long tm = tile / nGrp;

  int h = lane >> 4, r16 = lane & 15;
  int HoWo = Ho * Wo;
  long long mIdx = tm * 16 + r16;
  int nImg = (int)(mIdx / HoWo);
  int rem = (int)(mIdx % HoWo);
  int ho = rem / Wo, wo = rem % Wo;
  long long imgBase = (long long)nImg * H * W * CIN;
  int hb = ho * stride - pad, wb = wo * stride - pad;

  const int* wl = wp + (long long)(tg * NT) * kCh * 256 + lane * 8;

  v8i acc[NT];
  v8i zi = {0, 0, 0, 0, 0, 0, 0, 0};
#pragma unroll
  for (int j = 0; j < NT; ++j) acc[j] = zi;

#pragma unroll
  for (int kc = 0; kc < kCh; ++kc) {
    v8i A;
#pragma unroll
    for (int g = 0; g < 4; ++g) {
      int kb = kc * 64 + g * 16 + h * 8;
      int tap = kb / CIN, c = kb - tap * CIN;
      int ky = tap / KK, kx = tap - ky * KK;
      int y = hb + ky, x2 = wb + kx;
      bool inb = ((unsigned)y < (unsigned)H) & ((unsigned)x2 < (unsigned)W) & (kb < K);
      int yc = min(max(y, 0), H - 1), xc = min(max(x2, 0), W - 1);
      int2 dv = *(const int2*)(in + imgBase + ((long long)yc * W + xc) * CIN + c);
      A[2 * g] = inb ? dv.x : 0;
      A[2 * g + 1] = inb ? dv.y : 0;
    }
#pragma unroll
    for (int j = 0; j < NT; ++j) {
      v8i B = *(const v8i*)(wl + (long long)(j * kCh + kc) * 256);
      acc[j] = __builtin_amdgcn_wmma_i32_16x16x64_iu8(true, A, true, B, acc[j], false, false);
    }
  }

  long long mm0 = tm * 16 + 8 * h;
  int8_t* ob = out + mm0 * Cout;
#pragma unroll
  for (int j = 0; j < NT; ++j) {
    int ch = (tg * NT + j) * 16 + r16;
    float inv = bg[ch] * rsqrtf(bv[ch] + BN_EPS);
    float bias = bb[ch] - bm[ch] * inv;
#pragma unroll
    for (int r = 0; r < 8; ++r) {
      float valf = (float)acc[j][r] * inv + bias;
      ob[(long long)r * Cout + ch] = valf >= 0.f ? (int8_t)1 : (int8_t)-1;
    }
  }
}

template <int CIN, int KK, int NT>
__global__ __launch_bounds__(128) void hconv_t(
    const _Float16* __restrict__ in, void* __restrict__ outv,
    const _Float16* __restrict__ wp,
    const float* __restrict__ bg, const float* __restrict__ bb,
    const float* __restrict__ bm, const float* __restrict__ bv,
    unsigned* __restrict__ maxBits,
    int H, int W, int Cout, int stride, int pad, int Ho, int Wo, int mode)
{
  constexpr int K = CIN * KK * KK;
  constexpr int kCh = (K + 31) / 32;
  int lane = threadIdx.x & 31;
  int wave = threadIdx.x >> 5;
  long long M = 32ll * Ho * Wo;
  long long mTiles = M >> 4;
  int nGrp = (Cout >> 4) / NT;
  long long tile = (long long)blockIdx.x * 4 + wave;
  if (tile >= mTiles * nGrp) return;
  int tg = (int)(tile % nGrp);
  long long tm = tile / nGrp;

  int h = lane >> 4, r16 = lane & 15;
  int HoWo = Ho * Wo;
  long long mIdx = tm * 16 + r16;
  int nImg = (int)(mIdx / HoWo);
  int rem = (int)(mIdx % HoWo);
  int ho = rem / Wo, wo = rem % Wo;
  long long imgBase = (long long)nImg * H * W * CIN;
  int hb = ho * stride - pad, wb = wo * stride - pad;

  const _Float16* wl = wp + (long long)(tg * NT) * kCh * 512 + lane * 16;

  v8f acc[NT];
  v8f zf = {0.f, 0.f, 0.f, 0.f, 0.f, 0.f, 0.f, 0.f};
#pragma unroll
  for (int j = 0; j < NT; ++j) acc[j] = zf;

#pragma unroll
  for (int kc = 0; kc < kCh; ++kc) {
    v16h A;
#pragma unroll
    for (int g = 0; g < 2; ++g) {
      int kb = kc * 32 + h * 16 + g * 8;
      int tap = kb / CIN, c = kb - tap * CIN;
      int ky = tap / KK, kx = tap - ky * KK;
      int y = hb + ky, x2 = wb + kx;
      bool inb = ((unsigned)y < (unsigned)H) & ((unsigned)x2 < (unsigned)W) & (kb < K);
      int yc = min(max(y, 0), H - 1), xc = min(max(x2, 0), W - 1);
      v8h part = *(const v8h*)(in + imgBase + ((long long)yc * W + xc) * CIN + c);
      _Float16 zh = (_Float16)0.f;
#pragma unroll
      for (int e = 0; e < 8; ++e) A[g * 8 + e] = inb ? part[e] : zh;
    }
#pragma unroll
    for (int j = 0; j < NT; ++j) {
      v16h B = *(const v16h*)(wl + (long long)(j * kCh + kc) * 512);
      acc[j] = __builtin_amdgcn_wmma_f32_16x16x32_f16(false, A, false, B,
                                                      (short)0, acc[j], false, false);
    }
  }

  long long mm0 = tm * 16 + 8 * h;
  if (mode == 0) {
    float* pool = (float*)outv;
    int PH = Ho >> 1, PW = Wo >> 1;
    int ni0 = (int)(mm0 / HoWo);
    int rem0 = (int)(mm0 % HoWo);
    int oh0 = rem0 / Wo, ow0 = rem0 % Wo;
    float lmax = 0.f;
#pragma unroll
    for (int j = 0; j < NT; ++j) {
      int ch = (tg * NT + j) * 16 + r16;
      float inv = bg[ch] * rsqrtf(bv[ch] + BN_EPS);
      float bias = bb[ch] - bm[ch] * inv;
      int ni = ni0, oh = oh0, ow = ow0;
#pragma unroll
      for (int r = 0; r < 8; ++r) {
        float rl = fmaxf(acc[j][r] * inv + bias, 0.f);
        long long pidx = (((long long)ni * PH + (oh >> 1)) * PW + (ow >> 1)) * Cout + ch;
        atomicMax((unsigned*)(pool + pidx), __float_as_uint(rl));
        lmax = fmaxf(lmax, rl);
        if (++ow == Wo) { ow = 0; if (++oh == Ho) { oh = 0; ++ni; } }
      }
    }
    atomicMax(maxBits, __float_as_uint(lmax));
  } else {
    int8_t* ob = (int8_t*)outv + mm0 * Cout;
#pragma unroll
    for (int j = 0; j < NT; ++j) {
      int ch = (tg * NT + j) * 16 + r16;
      float inv = bg[ch] * rsqrtf(bv[ch] + BN_EPS);
      float bias = bb[ch] - bm[ch] * inv;
#pragma unroll
      for (int r = 0; r < 8; ++r) {
        float valf = acc[j][r] * inv + bias;
        ob[(long long)r * Cout + ch] = valf >= 0.f ? (int8_t)1 : (int8_t)-1;
      }
    }
  }
}

__global__ void pack_bin_kernel(const float* __restrict__ w, int* __restrict__ wp,
                                int Cin, int Cout, int k)
{
  int K = Cin * k * k, kCh = (K + 63) >> 6, nT = Cout >> 4;
  long long total = (long long)nT * kCh * 256;
  long long t = (long long)blockIdx.x * 256 + threadIdx.x;
  if (t >= total) return;
  int vj = (int)(t & 7);
  int lane = (int)((t >> 3) & 31);
  long long rest = t >> 8;
  int kc = (int)(rest % kCh);
  int tn = (int)(rest / kCh);
  int h = lane >> 4, n = tn * 16 + (lane & 15);
  unsigned dw = 0;
  for (int b2 = 0; b2 < 4; ++b2) {
    int kk = kc * 64 + (vj >> 1) * 16 + h * 8 + ((vj & 1) << 2) + b2;
    int bvv = 0;
    if (kk < K) {
      int tap = kk / Cin, c = kk - tap * Cin;
      int ky = tap / k, kx = tap - ky * k;
      float wv = w[(((long long)n * Cin + c) * k + ky) * k + kx];
      bvv = wv >= 0.f ? 1 : -1;
    }
    dw |= ((unsigned)(bvv & 0xFF)) << (8 * b2);
  }
  wp[t] = (int)dw;
}

__global__ void pack_w4_kernel(const float* __restrict__ w, _Float16* __restrict__ wp,
                               int CinR, int CinP, int Cout, int k)
{
  int K = CinP * k * k, kCh = (K + 31) >> 5, nT = Cout >> 4;
  long long total = (long long)nT * kCh * 256;
  long long t = (long long)blockIdx.x * 256 + threadIdx.x;
  if (t >= total) return;
  int vj = (int)(t & 7);
  int lane = (int)((t >> 3) & 31);
  long long rest = t >> 8;
  int kc = (int)(rest % kCh);
  int tn = (int)(rest / kCh);
  int h = lane >> 4, n = tn * 16 + (lane & 15);
  int cnt = CinR * k * k;
  float ma = 0.f;
  for (int i = 0; i < cnt; ++i) ma = fmaxf(ma, fabsf(w[(long long)n * cnt + i]));
  float s = exp2f(ceilf(log2f(ma / 7.f + 1e-12f)));
  for (int e = 0; e < 2; ++e) {
    int kk = kc * 32 + h * 16 + vj * 2 + e;
    float q = 0.f;
    if (kk < K) {
      int tap = kk / CinP, c = kk - tap * CinP;
      if (c < CinR) {
        int ky = tap / k, kx = tap - ky * k;
        float wv = w[(((long long)n * CinR + c) * k + ky) * k + kx];
        q = fminf(fmaxf(roundf(wv / s), -7.f), 7.f) * s;
      }
    }
    wp[t * 2 + e] = (_Float16)q;
  }
}

__global__ void prep_in_kernel(const float* __restrict__ x, _Float16* __restrict__ out,
                               long long total)
{
  long long i = (long long)blockIdx.x * 256 + threadIdx.x;
  if (i >= total) return;
  long long n = i / (224 * 224);
  int rem = (int)(i % (224 * 224));
#pragma unroll
  for (int c = 0; c < 16; ++c) {
    float q = 0.f;
    if (c < 3) {
      float xv = x[(n * 3 + c) * (224 * 224) + rem];
      q = fmaxf(fminf(roundf(xv * 128.f), 127.f), -128.f) * 0.0078125f;
    }
    out[i * 16 + c] = (_Float16)q;
  }
}

__global__ void zero_kernel(unsigned* __restrict__ p, long long n)
{
  long long i = (long long)blockIdx.x * 256 + threadIdx.x;
  if (i < n) p[i] = 0u;
}

__global__ void quantpool_kernel(const float* __restrict__ pool,
                                 const unsigned* __restrict__ maxBits,
                                 _Float16* __restrict__ out, long long total)
{
  long long i = (long long)blockIdx.x * 256 + threadIdx.x;
  if (i >= total) return;
  float mx = __uint_as_float(*maxBits);
  float s = exp2f(ceilf(log2f(mx / 15.f + 1e-12f)));
  float q = fminf(fmaxf(roundf(pool[i] / s), 0.f), 15.f) * s;
  out[i] = (_Float16)q;
}

__global__ void add8_kernel(const int8_t* __restrict__ a, const int8_t* __restrict__ b,
                            int8_t* __restrict__ c, long long n)
{
  long long i = (long long)blockIdx.x * 256 + threadIdx.x;
  if (i < n) c[i] = (int8_t)(a[i] + b[i]);
}

__global__ void head_kernel(const int8_t* __restrict__ x, const float* __restrict__ fw,
                            const float* __restrict__ fb, float* __restrict__ out)
{
  __shared__ float mean[128];
  int n = blockIdx.x, c = threadIdx.x;
  int s = 0;
  for (int p = 0; p < 49; ++p) s += x[((long long)n * 49 + p) * 128 + c];
  mean[c] = (float)s / 49.f;
  __syncthreads();
  if (c < 2) {
    float a = fb[c];
    for (int j = 0; j < 128; ++j) a += mean[j] * fw[c * 128 + j];
    out[n * 2 + c] = a;
  }
}

extern "C" void kernel_launch(void* const* d_in, const int* in_sizes, int n_in,
                              void* d_out, int out_size, void* d_ws, size_t ws_size,
                              hipStream_t stream) {
  (void)in_sizes; (void)out_size; (void)ws_size;
  if (n_in < 103) return;

  const float* x = (const float*)d_in[0];
  auto Wt = [&](int i) { return (const float*)d_in[1 + i]; };
  auto BN = [&](int i, int j) { return (const float*)d_in[21 + 4 * i + j]; };
  const float* fw = (const float*)d_in[101];
  const float* fb = (const float*)d_in[102];
  float* out = (float*)d_out;

  static const int SPEC[20][5] = {
      {3, 64, 3, 1, 1},  {64, 32, 3, 2, 1},
      {32, 16, 1, 1, 0}, {16, 32, 3, 1, 1}, {32, 32, 1, 1, 0},
      {32, 64, 3, 2, 1},
      {64, 32, 1, 1, 0}, {32, 64, 3, 1, 1}, {64, 32, 1, 1, 0},
      {32, 64, 3, 1, 1}, {64, 64, 1, 1, 0},
      {64, 96, 3, 2, 1},
      {96, 48, 1, 1, 0}, {48, 96, 3, 1, 1}, {96, 48, 1, 1, 0},
      {48, 96, 3, 1, 1}, {96, 96, 3, 1, 1},
      {96, 128, 3, 2, 1},
      {128, 128, 3, 1, 1}, {128, 128, 3, 1, 1}};

  char* ws = (char*)d_ws;
  const size_t SZ_IN16 = 32ull * 224 * 224 * 16 * 2;
  const size_t SZ_POOL = 32ull * 112 * 112 * 64 * 4;
  size_t off = 0;
  _Float16* bufIn16 = (_Float16*)(ws + off); off += SZ_IN16;
  float* poolBuf = (float*)(ws + off);       off += SZ_POOL;
  unsigned* maxBits = (unsigned*)(ws + off); off += 256;
  int8_t* bufA = (int8_t*)(ws + off); off += (size_t)4 << 20;
  int8_t* bufB = (int8_t*)(ws + off); off += (size_t)4 << 20;
  int8_t* bufC = (int8_t*)(ws + off); off += (size_t)4 << 20;
  void* wpack[20];
  for (int i = 0; i < 20; ++i) {
    int cin = SPEC[i][0], cout = SPEC[i][1], k = SPEC[i][2];
    size_t sz;
    if (i < 2) {
      int cp = (i == 0) ? 16 : cin;
      int kCh = (cp * k * k + 31) >> 5;
      sz = (size_t)(cout / 16) * kCh * 1024;
    } else {
      int kCh = (cin * k * k + 63) >> 6;
      sz = (size_t)(cout / 16) * kCh * 1024;
    }
    wpack[i] = (void*)(ws + off);
    off += (sz + 255) & ~(size_t)255;
  }

  long long nPix = 32ll * 224 * 224;
  prep_in_kernel<<<(unsigned)((nPix + 255) / 256), 256, 0, stream>>>(x, bufIn16, nPix);
  long long nz = (long long)(SZ_POOL / 4) + 64;
  zero_kernel<<<(unsigned)((nz + 255) / 256), 256, 0, stream>>>((unsigned*)poolBuf, nz);

  {
    long long t0 = (long long)(64 / 16) * ((16 * 9 + 31) >> 5) * 256;
    pack_w4_kernel<<<(unsigned)((t0 + 255) / 256), 256, 0, stream>>>(
        Wt(0), (_Float16*)wpack[0], 3, 16, 64, 3);
    long long t1 = (long long)(32 / 16) * ((64 * 9 + 31) >> 5) * 256;
    pack_w4_kernel<<<(unsigned)((t1 + 255) / 256), 256, 0, stream>>>(
        Wt(1), (_Float16*)wpack[1], 64, 64, 32, 3);
    for (int i = 2; i < 20; ++i) {
      int cin = SPEC[i][0], cout = SPEC[i][1], k = SPEC[i][2];
      long long tt = (long long)(cout / 16) * ((cin * k * k + 63) >> 6) * 256;
      pack_bin_kernel<<<(unsigned)((tt + 255) / 256), 256, 0, stream>>>(
          Wt(i), (int*)wpack[i], cin, cout, k);
    }
  }

#define LAUNCH_H(li, CINv, KKv, NTv, inp, outp, Hv, Wv, modev)                      \
  do {                                                                              \
    int cout = SPEC[li][1], k = SPEC[li][2], s = SPEC[li][3], p = SPEC[li][4];      \
    int Ho = (Hv + 2 * p - k) / s + 1, Wo = (Wv + 2 * p - k) / s + 1;               \
    long long tiles = ((32ll * Ho * Wo) / 16) * ((cout / 16) / NTv);                \
    hconv_t<CINv, KKv, NTv><<<(unsigned)((tiles + 3) / 4), 128, 0, stream>>>(       \
        inp, outp, (const _Float16*)wpack[li], BN(li, 0), BN(li, 1), BN(li, 2),     \
        BN(li, 3), maxBits, Hv, Wv, cout, s, p, Ho, Wo, modev);                     \
  } while (0)

#define LAUNCH_B(li, CINv, KKv, NTv, inp, outp, Hv, Wv)                             \
  do {                                                                              \
    int cout = SPEC[li][1], k = SPEC[li][2], s = SPEC[li][3], p = SPEC[li][4];      \
    int Ho = (Hv + 2 * p - k) / s + 1, Wo = (Wv + 2 * p - k) / s + 1;               \
    long long tiles = ((32ll * Ho * Wo) / 16) * ((cout / 16) / NTv);                \
    binconv_t<CINv, KKv, NTv><<<(unsigned)((tiles + 3) / 4), 128, 0, stream>>>(     \
        inp, outp, (const int*)wpack[li], BN(li, 0), BN(li, 1), BN(li, 2),          \
        BN(li, 3), Hv, Wv, cout, s, p, Ho, Wo);                                     \
  } while (0)

  LAUNCH_H(0, 16, 3, 4, bufIn16, (void*)poolBuf, 224, 224, 0);
  _Float16* bufP = bufIn16;
  long long nPool = 32ll * 112 * 112 * 64;
  quantpool_kernel<<<(unsigned)((nPool + 255) / 256), 256, 0, stream>>>(
      poolBuf, maxBits, bufP, nPool);
  LAUNCH_H(1, 64, 3, 2, bufP, (void*)bufA, 112, 112, 1);

  LAUNCH_B(2, 32, 1, 1, bufA, bufB, 56, 56);
  LAUNCH_B(3, 16, 3, 2, bufB, bufC, 56, 56);
  LAUNCH_B(4, 32, 1, 2, bufC, bufB, 56, 56);
  long long n1 = 32ll * 56 * 56 * 32;
  add8_kernel<<<(unsigned)((n1 + 255) / 256), 256, 0, stream>>>(bufA, bufB, bufC, n1);
  LAUNCH_B(5, 32, 3, 4, bufC, bufA, 56, 56);
  LAUNCH_B(6, 64, 1, 2, bufA, bufB, 28, 28);
  LAUNCH_B(7, 32, 3, 4, bufB, bufC, 28, 28);
  LAUNCH_B(8, 64, 1, 2, bufC, bufB, 28, 28);
  LAUNCH_B(9, 32, 3, 4, bufB, bufC, 28, 28);
  LAUNCH_B(10, 64, 1, 4, bufC, bufB, 28, 28);
  long long n2 = 32ll * 28 * 28 * 64;
  add8_kernel<<<(unsigned)((n2 + 255) / 256), 256, 0, stream>>>(bufA, bufB, bufC, n2);
  LAUNCH_B(11, 64, 3, 3, bufC, bufA, 28, 28);
  LAUNCH_B(12, 96, 1, 3, bufA, bufB, 14, 14);
  LAUNCH_B(13, 48, 3, 3, bufB, bufC, 14, 14);
  LAUNCH_B(14, 96, 1, 3, bufC, bufB, 14, 14);
  LAUNCH_B(15, 48, 3, 3, bufB, bufC, 14, 14);
  LAUNCH_B(16, 96, 3, 3, bufC, bufB, 14, 14);
  long long n3 = 32ll * 14 * 14 * 96;
  add8_kernel<<<(unsigned)((n3 + 255) / 256), 256, 0, stream>>>(bufA, bufB, bufC, n3);
  LAUNCH_B(17, 96, 3, 4, bufC, bufA, 14, 14);
  LAUNCH_B(18, 128, 3, 4, bufA, bufB, 7, 7);
  LAUNCH_B(19, 128, 3, 4, bufB, bufC, 7, 7);
  long long n4 = 32ll * 7 * 7 * 128;
  add8_kernel<<<(unsigned)((n4 + 255) / 256), 256, 0, stream>>>(bufA, bufC, bufB, n4);

  head_kernel<<<32, 128, 0, stream>>>(bufB, fw, fb, out);

#undef LAUNCH_H
#undef LAUNCH_B
}
